// CopyRnnTF_90108413870703
// MI455X (gfx1250) — compile-verified, matrix operand order checked by reading, not run on hardware
//
#include <hip/hip_runtime.h>
#include <hip/hip_bf16.h>
#include <math.h>

typedef __attribute__((ext_vector_type(16))) _Float16 v16h;
typedef __attribute__((ext_vector_type(8)))  _Float16 v8h;
typedef __attribute__((ext_vector_type(8)))  float    v8f;

namespace {
constexpr int kV = 50000, kOOV = 100, kVO = 50100;
constexpr int kE = 128, kH = 256, kG = 1024;     // kH = HE = HD, kG = 4*kH
constexpr int kB = 32, kS = 256, kT = 16;
}

// ---------------------------------------------------------------------------
// WMMA 16-bit operand fragment load (CDNA5 wave32 layout):
//   lanes 0-15 : row = lane,    halves 0-7 -> K=k0..k0+7,  8-15 -> K=k0+16..k0+23
//   lanes 16-31: row = lane-16, halves 0-7 -> K=k0+8..+15, 8-15 -> K=k0+24..+31
// p -> fragment origin (row0, k0); rows contiguous, leading dim ld (mult of 8).
// ---------------------------------------------------------------------------
__device__ __forceinline__ v16h load_frag(const _Float16* __restrict__ p,
                                          int ld, int lane) {
  int r = lane & 15, h = (lane >> 4) & 1;
  const _Float16* q = p + r * ld + h * 8;
  v8h lo = *(const v8h*)(q);
  v8h hi = *(const v8h*)(q + 16);
  v16h out;
#pragma unroll
  for (int j = 0; j < 8; ++j) { out[j] = lo[j]; out[8 + j] = hi[j]; }
  return out;
}

// ---------------------------------------------------------------------------
// Batched WMMA GEMM:  C[b] = act( A[b] @ B[b]^T + Cadd[b] + bias )
//   A: M x K row-major f16.   B: N x K row-major f16 (pre-transposed weights).
//   One wave computes a 16 x (16*NT) strip: 1 A-fragment feeds NT independent
//   accumulators (A reuse + breaks WMMA->WMMA RAW hazard chains).
//   Remainder handling: overlap-clamp (requires N/16 >= NT, true for all our
//   GEMMs) -> single branch-free inner loop, compile-time fragment indices.
//   K-loop: 2x-unrolled ping-pong software pipeline — prefetch step k+1 into
//   buffer 1 while computing on buffer 0, then prefetch k+2 into buffer 0
//   while computing on buffer 1. Fixed registers, no rotation copies.
// ---------------------------------------------------------------------------
template <int NT>
__global__ __launch_bounds__(128) void wmma_gemm_kernel(
    const _Float16* __restrict__ A, int lda, long long strideA,
    const _Float16* __restrict__ Bm, int ldb, long long strideB,
    float* __restrict__ C, long long ldc, long long strideC,
    _Float16* __restrict__ C16, int ldc16,
    const float* __restrict__ Cadd, int ldadd, long long strideAdd,
    const float* __restrict__ bias,
    int M, int N, int K, int act) {
  int lane = threadIdx.x & 31;
  int wid  = blockIdx.x * (blockDim.x >> 5) + (threadIdx.x >> 5);
  int ntN = N >> 4, ntM = M >> 4;
  int groups = (ntN + NT - 1) / NT;
  if (wid >= ntM * groups) return;
  int mt = wid / groups, ng = wid % groups;
  int nt0 = ng * NT;
  if (nt0 > ntN - NT) nt0 = ntN - NT;   // overlap-clamp: no remainder path

  long long batch = blockIdx.z;
  A  += batch * strideA;
  Bm += batch * strideB;
  C  += batch * strideC;
  if (Cadd) Cadd += batch * strideAdd;

  int ln = lane & 15, lh = (lane >> 4) & 1;

  v8f acc[NT];
#pragma unroll
  for (int u = 0; u < NT; ++u) {
    int colC = (nt0 + u) * 16 + ln;
#pragma unroll
    for (int v = 0; v < 8; ++v) {
      int row = mt * 16 + v + 8 * lh;
      float xv = 0.f;
      if (Cadd) xv += Cadd[(long long)row * ldadd + colC];
      if (bias) xv += bias[colC];
      acc[u][v] = xv;
    }
  }

  const _Float16* pa = A + (long long)(mt * 16) * lda;
  const _Float16* pb = Bm + (long long)(nt0 * 16) * ldb;

  // ---- 2x-unrolled ping-pong pipelined K loop ----
  v16h a0, a1;
  v16h b0[NT], b1[NT];

  a0 = load_frag(pa, lda, lane);
#pragma unroll
  for (int u = 0; u < NT; ++u)
    b0[u] = load_frag(pb + (long long)(u * 16) * ldb, ldb, lane);

  int k0 = 0;
  for (; k0 + 64 < K; k0 += 64) {
    // prefetch step k0+32 into buffer 1
    a1 = load_frag(pa + k0 + 32, lda, lane);
#pragma unroll
    for (int u = 0; u < NT; ++u)
      b1[u] = load_frag(pb + (long long)(u * 16) * ldb + k0 + 32, ldb, lane);
    // compute step k0 from buffer 0
#pragma unroll
    for (int u = 0; u < NT; ++u)
      acc[u] = __builtin_amdgcn_wmma_f32_16x16x32_f16(false, a0, false, b0[u],
                                                      (short)0, acc[u], false, false);
    // prefetch step k0+64 into buffer 0
    a0 = load_frag(pa + k0 + 64, lda, lane);
#pragma unroll
    for (int u = 0; u < NT; ++u)
      b0[u] = load_frag(pb + (long long)(u * 16) * ldb + k0 + 64, ldb, lane);
    // compute step k0+32 from buffer 1
#pragma unroll
    for (int u = 0; u < NT; ++u)
      acc[u] = __builtin_amdgcn_wmma_f32_16x16x32_f16(false, a1, false, b1[u],
                                                      (short)0, acc[u], false, false);
  }
  if (k0 + 32 < K) {
    // two steps remain: k0 (in buffer 0) and k0+32
    a1 = load_frag(pa + k0 + 32, lda, lane);
#pragma unroll
    for (int u = 0; u < NT; ++u)
      b1[u] = load_frag(pb + (long long)(u * 16) * ldb + k0 + 32, ldb, lane);
#pragma unroll
    for (int u = 0; u < NT; ++u)
      acc[u] = __builtin_amdgcn_wmma_f32_16x16x32_f16(false, a0, false, b0[u],
                                                      (short)0, acc[u], false, false);
#pragma unroll
    for (int u = 0; u < NT; ++u)
      acc[u] = __builtin_amdgcn_wmma_f32_16x16x32_f16(false, a1, false, b1[u],
                                                      (short)0, acc[u], false, false);
  } else {
    // one step remains (buffer 0)
#pragma unroll
    for (int u = 0; u < NT; ++u)
      acc[u] = __builtin_amdgcn_wmma_f32_16x16x32_f16(false, a0, false, b0[u],
                                                      (short)0, acc[u], false, false);
  }

#pragma unroll
  for (int u = 0; u < NT; ++u) {
    int colC = (nt0 + u) * 16 + ln;
#pragma unroll
    for (int v = 0; v < 8; ++v) {
      int row = mt * 16 + v + 8 * lh;
      float x = acc[u][v];
      if (act == 1) x = tanhf(x);
      else if (act == 2) x = expf(x);
      C[(long long)row * ldc + colC] = x;
      if (C16) C16[(long long)row * ldc16 + colC] = (_Float16)x;
    }
  }
}

// ---------------------------------------------------------------------------
// Elementwise / utility kernels
// ---------------------------------------------------------------------------
__global__ void fill_f32_kernel(float* p, float v, int n) {
  int i = blockIdx.x * blockDim.x + threadIdx.x;
  if (i < n) p[i] = v;
}
__global__ void fill_f16_kernel(_Float16* p, float v, int n) {
  int i = blockIdx.x * blockDim.x + threadIdx.x;
  if (i < n) p[i] = (_Float16)v;
}
__global__ void cvt_f16_kernel(const float* __restrict__ in, _Float16* __restrict__ out, int n) {
  int i = blockIdx.x * blockDim.x + threadIdx.x;
  if (i < n) out[i] = (_Float16)in[i];
}
__global__ void copy_f32_kernel(const float* __restrict__ in, float* __restrict__ out, int n) {
  int i = blockIdx.x * blockDim.x + threadIdx.x;
  if (i < n) out[i] = in[i];
}
__global__ void embed_kernel(const int* __restrict__ tok, const float* __restrict__ emb,
                             _Float16* __restrict__ out, int ntok, int E) {
  int i = blockIdx.x * blockDim.x + threadIdx.x;
  if (i >= ntok * E) return;
  int t = i / E, e = i % E;
  out[i] = (_Float16)emb[(long long)tok[t] * E + e];
}
__global__ void transpose_f16_kernel(const float* __restrict__ in, _Float16* __restrict__ out,
                                     int K, int N) {
  long long i = (long long)blockIdx.x * blockDim.x + threadIdx.x;
  if (i >= (long long)K * N) return;
  int k = (int)(i / N), n = (int)(i % N);
  out[(long long)n * K + k] = (_Float16)in[i];
}
// (B,S,E) f32 -> (B,E,S) f16
__global__ void transpose_batched_f16_kernel(const float* __restrict__ in,
                                             _Float16* __restrict__ out,
                                             int S, int E, int n) {
  int i = blockIdx.x * blockDim.x + threadIdx.x;
  if (i >= n) return;
  int e = i % E, rest = i / E;
  int s = rest % S, b = rest / S;
  out[((long long)b * E + e) * S + s] = (_Float16)in[i];
}

// LSTM cell pointwise: z = [i f g o] gates (B x 4H); updates c,h and emits seq
__global__ void lstm_pointwise_kernel(const float* __restrict__ z, int H,
                                      float* __restrict__ c, float* __restrict__ h,
                                      _Float16* __restrict__ h16,
                                      float* __restrict__ seq, _Float16* __restrict__ seq16,
                                      int seq_stride, int n) {
  int i = blockIdx.x * blockDim.x + threadIdx.x;
  if (i >= n) return;
  int b = i / H, j = i % H;
  const float* zb = z + (long long)b * 4 * H;
  float ig = zb[j], fg = zb[H + j], gg = zb[2 * H + j], og = zb[3 * H + j];
  float si = 1.f / (1.f + expf(-ig));
  float sf = 1.f / (1.f + expf(-fg));
  float so = 1.f / (1.f + expf(-og));
  float cn = sf * c[i] + si * tanhf(gg);
  float hn = so * tanhf(cn);
  c[i] = cn; h[i] = hn; h16[i] = (_Float16)hn;
  if (seq) {
    seq[(long long)b * seq_stride + j] = hn;
    seq16[(long long)b * seq_stride + j] = (_Float16)hn;
  }
}

// masked softmax over S=256; one block (256 thr) per (b,t) row
__global__ void softmax_mask_kernel(const float* __restrict__ scores,
                                    const int* __restrict__ xlen,
                                    _Float16* __restrict__ aw16, int S, int T) {
  int row = blockIdx.x;          // b*T + t
  int b = row / T;
  int len = xlen[b];
  int s = threadIdx.x;
  float v = (s < len) ? scores[(long long)row * S + s] : -1e20f;
  __shared__ float red[256];
  red[s] = v; __syncthreads();
  for (int off = 128; off > 0; off >>= 1) {
    if (s < off) red[s] = fmaxf(red[s], red[s + off]);
    __syncthreads();
  }
  float m = red[0]; __syncthreads();
  float e = expf(v - m);
  red[s] = e; __syncthreads();
  for (int off = 128; off > 0; off >>= 1) {
    if (s < off) red[s] += red[s + off];
    __syncthreads();
  }
  aw16[(long long)row * S + s] = (_Float16)(e / red[0]);
}

// ctx/dec concat -> f16 (B*T x 2H)
__global__ void concat_kernel(const float* __restrict__ ctx, const float* __restrict__ dec,
                              _Float16* __restrict__ cat, int n) {
  int i = blockIdx.x * blockDim.x + threadIdx.x;
  if (i >= n) return;
  int row = i / (2 * kH), j = i % (2 * kH);
  float v = (j < kH) ? ctx[(long long)row * kH + j] : dec[(long long)row * kH + (j - kH)];
  cat[i] = (_Float16)v;
}

__global__ void init_oov_kernel(float* __restrict__ out, int n) {
  int i = blockIdx.x * blockDim.x + threadIdx.x;
  if (i >= n) return;
  int bt = i / kOOV, j = i % kOOV;
  out[(long long)bt * kVO + kV + j] = 1e-10f;
}

// deterministic copy-distribution scatter: one thread per (b,t) row
__global__ void copy_scatter_kernel(const float* __restrict__ cs, const int* __restrict__ xoov,
                                    const int* __restrict__ xlen, float* __restrict__ out) {
  int bt = blockIdx.x * blockDim.x + threadIdx.x;
  if (bt >= kB * kT) return;
  int b = bt / kT;
  int len = xlen[b];
  float* row = out + (long long)bt * kVO;
  const float* c = cs + (long long)bt * kS;
  const int* ids = xoov + (long long)b * kS;
  for (int s = 0; s < len; ++s) row[ids[s]] += expf(c[s]);
}

// per-(b,t) row: log(total / sum(total)) in place
__global__ void finalize_kernel(float* __restrict__ out) {
  int row = blockIdx.x;
  float* p = out + (long long)row * kVO;
  __shared__ float red[256];
  float s = 0.f;
  for (int i = threadIdx.x; i < kVO; i += 256) s += p[i];
  red[threadIdx.x] = s; __syncthreads();
  for (int off = 128; off > 0; off >>= 1) {
    if (threadIdx.x < off) red[threadIdx.x] += red[threadIdx.x + off];
    __syncthreads();
  }
  float ls = logf(red[0]);
  for (int i = threadIdx.x; i < kVO; i += 256) p[i] = logf(p[i]) - ls;
}

// ---------------------------------------------------------------------------
// Host-side helpers
// ---------------------------------------------------------------------------
static inline int cdiv(int a, int b) { return (a + b - 1) / b; }

static void launch_gemm(hipStream_t st,
                        const _Float16* A, int lda, long long sA,
                        const _Float16* Bm, int ldb, long long sB,
                        float* C, long long ldc, long long sC,
                        _Float16* C16, int ldc16,
                        const float* Cadd, int ldadd, long long sAdd,
                        const float* bias,
                        int M, int N, int K, int act, int nbatch) {
  constexpr int NT = 4;
  int groups = cdiv(N / 16, NT);
  int tiles = (M / 16) * groups;
  dim3 grid(cdiv(tiles, 4), 1, nbatch);
  wmma_gemm_kernel<NT><<<grid, 128, 0, st>>>(A, lda, sA, Bm, ldb, sB, C, ldc, sC,
                                             C16, ldc16, Cadd, ldadd, sAdd, bias,
                                             M, N, K, act);
}

extern "C" void kernel_launch(void* const* d_in, const int* in_sizes, int n_in,
                              void* d_out, int out_size, void* d_ws, size_t ws_size,
                              hipStream_t stream) {
  (void)in_sizes; (void)n_in; (void)out_size; (void)ws_size;
  const int*   x     = (const int*)d_in[0];
  const int*   xoov  = (const int*)d_in[1];
  const int*   xlen  = (const int*)d_in[2];
  const int*   decx  = (const int*)d_in[3];
  const float* emb   = (const float*)d_in[4];
  const float* Wenc  = (const float*)d_in[5];
  const float* Uenc  = (const float*)d_in[6];
  const float* benc  = (const float*)d_in[7];
  const float* We2d  = (const float*)d_in[8];
  const float* be2d  = (const float*)d_in[9];
  const float* Wdec  = (const float*)d_in[10];
  const float* Udec  = (const float*)d_in[11];
  const float* bdec  = (const float*)d_in[12];
  const float* Wattn = (const float*)d_in[13];
  const float* Wout  = (const float*)d_in[14];
  const float* bout  = (const float*)d_in[15];
  const float* Wgen  = (const float*)d_in[16];
  const float* Wcopy = (const float*)d_in[17];
  const float* bcopy = (const float*)d_in[18];
  float* out = (float*)d_out;

  // ---- workspace bump allocator ----
  size_t off = 0;
  auto alloc = [&](size_t bytes) -> void* {
    void* p = (char*)d_ws + off;
    off += (bytes + 255) & ~(size_t)255;
    return p;
  };
  _Float16* WencT  = (_Float16*)alloc((size_t)kG * kE * 2);
  _Float16* UencT  = (_Float16*)alloc((size_t)kG * kH * 2);
  _Float16* WdecT  = (_Float16*)alloc((size_t)kG * kE * 2);
  _Float16* UdecT  = (_Float16*)alloc((size_t)kG * kH * 2);
  _Float16* We2dT  = (_Float16*)alloc((size_t)kH * kH * 2);
  _Float16* WattnT = (_Float16*)alloc((size_t)kH * kH * 2);
  _Float16* WoutT  = (_Float16*)alloc((size_t)kH * 2 * kH * 2);
  _Float16* WcopyT = (_Float16*)alloc((size_t)kH * kH * 2);
  _Float16* WgenT  = (_Float16*)alloc((size_t)kV * kH * 2);       // 25.6 MB
  _Float16* Aenc   = (_Float16*)alloc((size_t)kB * kS * kE * 2);
  _Float16* Adec   = (_Float16*)alloc((size_t)kB * kT * kE * 2);
  float*    xWenc  = (float*)alloc((size_t)kB * kS * kG * 4);     // 33.5 MB
  float*    xWdec  = (float*)alloc((size_t)kB * kT * kG * 4);
  float*    zbuf   = (float*)alloc((size_t)kB * kG * 4);
  float*    h_enc  = (float*)alloc((size_t)kB * kH * 4);
  float*    c_enc  = (float*)alloc((size_t)kB * kH * 4);
  _Float16* h_enc16 = (_Float16*)alloc((size_t)kB * kH * 2);
  _Float16* c_enc16 = (_Float16*)alloc((size_t)kB * kH * 2);
  float*    enc_out   = (float*)alloc((size_t)kB * kS * kH * 4);  // 8 MB
  _Float16* enc_out16 = (_Float16*)alloc((size_t)kB * kS * kH * 2);
  _Float16* enc_outT16 = (_Float16*)alloc((size_t)kB * kH * kS * 2); // (B,E,S)
  float*    h_dec  = (float*)alloc((size_t)kB * kH * 4);
  float*    c_dec  = (float*)alloc((size_t)kB * kH * 4);
  _Float16* h_dec16 = (_Float16*)alloc((size_t)kB * kH * 2);
  float*    dec_out   = (float*)alloc((size_t)kB * kT * kH * 4);
  _Float16* dec_out16 = (_Float16*)alloc((size_t)kB * kT * kH * 2);
  float*    qbuf   = (float*)alloc((size_t)kB * kT * kH * 4);
  _Float16* q16    = (_Float16*)alloc((size_t)kB * kT * kH * 2);
  float*    scores = (float*)alloc((size_t)kB * kT * kS * 4);
  _Float16* aw16   = (_Float16*)alloc((size_t)kB * kT * kS * 2);
  float*    ctx    = (float*)alloc((size_t)kB * kT * kH * 4);
  _Float16* cat16  = (_Float16*)alloc((size_t)kB * kT * 2 * kH * 2);
  float*    attn   = (float*)alloc((size_t)kB * kT * kH * 4);
  _Float16* attn16 = (_Float16*)alloc((size_t)kB * kT * kH * 2);
  float*    encproj   = (float*)alloc((size_t)kB * kS * kH * 4);  // 8 MB
  _Float16* encproj16 = (_Float16*)alloc((size_t)kB * kS * kH * 2);
  float*    cs     = (float*)alloc((size_t)kB * kT * kS * 4);

  // ---- weight transposes -> f16 N-major (WMMA-friendly contiguous K) ----
  auto tr = [&](const float* in, _Float16* o, int K, int N) {
    long long n = (long long)K * N;
    transpose_f16_kernel<<<(unsigned)((n + 255) / 256), 256, 0, stream>>>(in, o, K, N);
  };
  tr(Wenc,  WencT,  kE, kG);
  tr(Uenc,  UencT,  kH, kG);
  tr(Wdec,  WdecT,  kE, kG);
  tr(Udec,  UdecT,  kH, kG);
  tr(We2d,  We2dT,  kH, kH);
  tr(Wattn, WattnT, kH, kH);
  tr(Wout,  WoutT,  2 * kH, kH);
  tr(Wcopy, WcopyT, kH, kH);
  tr(Wgen,  WgenT,  kH, kV);

  // ---- embedding gathers ----
  embed_kernel<<<cdiv(kB * kS * kE, 256), 256, 0, stream>>>(x, emb, Aenc, kB * kS, kE);
  embed_kernel<<<cdiv(kB * kT * kE, 256), 256, 0, stream>>>(decx, emb, Adec, kB * kT, kE);

  // ---- hoisted input projections: xW = emb(x) @ W + b ----
  launch_gemm(stream, Aenc, kE, 0, WencT, kE, 0, xWenc, kG, 0, nullptr, 0,
              nullptr, 0, 0, benc, kB * kS, kG, kE, 0, 1);
  launch_gemm(stream, Adec, kE, 0, WdecT, kE, 0, xWdec, kG, 0, nullptr, 0,
              nullptr, 0, 0, bdec, kB * kT, kG, kE, 0, 1);

  // ---- encoder LSTM ----
  fill_f32_kernel<<<cdiv(kB * kH, 256), 256, 0, stream>>>(h_enc, 0.f, kB * kH);
  fill_f32_kernel<<<cdiv(kB * kH, 256), 256, 0, stream>>>(c_enc, 0.f, kB * kH);
  fill_f16_kernel<<<cdiv(kB * kH, 256), 256, 0, stream>>>(h_enc16, 0.f, kB * kH);
  for (int t = 0; t < kS; ++t) {
    launch_gemm(stream, h_enc16, kH, 0, UencT, kH, 0, zbuf, kG, 0, nullptr, 0,
                xWenc + (long long)t * kG, kS * kG, 0, nullptr,
                kB, kG, kH, 0, 1);
    lstm_pointwise_kernel<<<cdiv(kB * kH, 256), 256, 0, stream>>>(
        zbuf, kH, c_enc, h_enc, h_enc16,
        enc_out + (long long)t * kH, enc_out16 + (long long)t * kH, kS * kH, kB * kH);
  }

  // ---- bridge: h0/c0 = {eh,ec} @ We2d + be2d ----
  cvt_f16_kernel<<<cdiv(kB * kH, 256), 256, 0, stream>>>(c_enc, c_enc16, kB * kH);
  launch_gemm(stream, h_enc16, kH, 0, We2dT, kH, 0, h_dec, kH, 0, h_dec16, kH,
              nullptr, 0, 0, be2d, kB, kH, kH, 0, 1);
  launch_gemm(stream, c_enc16, kH, 0, We2dT, kH, 0, c_dec, kH, 0, nullptr, 0,
              nullptr, 0, 0, be2d, kB, kH, kH, 0, 1);

  // ---- decoder LSTM ----
  for (int t = 0; t < kT; ++t) {
    launch_gemm(stream, h_dec16, kH, 0, UdecT, kH, 0, zbuf, kG, 0, nullptr, 0,
                xWdec + (long long)t * kG, kT * kG, 0, nullptr,
                kB, kG, kH, 0, 1);
    lstm_pointwise_kernel<<<cdiv(kB * kH, 256), 256, 0, stream>>>(
        zbuf, kH, c_dec, h_dec, h_dec16,
        dec_out + (long long)t * kH, dec_out16 + (long long)t * kH, kT * kH, kB * kH);
  }

  // ---- attention: q = dec_out @ Wattn ; scores = q . enc_out ----
  launch_gemm(stream, dec_out16, kH, 0, WattnT, kH, 0, qbuf, kH, 0, q16, kH,
              nullptr, 0, 0, nullptr, kB * kT, kH, kH, 0, 1);
  launch_gemm(stream, q16, kH, (long long)kT * kH, enc_out16, kH, (long long)kS * kH,
              scores, kS, (long long)kT * kS, nullptr, 0, nullptr, 0, 0, nullptr,
              kT, kS, kH, 0, kB);
  softmax_mask_kernel<<<kB * kT, kS, 0, stream>>>(scores, xlen, aw16, kS, kT);

  // ---- ctx = aw @ enc_out via pre-transposed enc_out (B,E,S) f16 ----
  transpose_batched_f16_kernel<<<cdiv(kB * kS * kH, 256), 256, 0, stream>>>(
      enc_out, enc_outT16, kS, kH, kB * kS * kH);
  launch_gemm(stream, aw16, kS, (long long)kT * kS, enc_outT16, kS, (long long)kH * kS,
              ctx, kH, (long long)kT * kH, nullptr, 0, nullptr, 0, 0, nullptr,
              kT, kH, kS, 0, kB);

  // ---- attn_out = tanh([ctx, dec_out] @ Wout + bout) ----
  concat_kernel<<<cdiv(kB * kT * 2 * kH, 256), 256, 0, stream>>>(ctx, dec_out, cat16,
                                                                 kB * kT * 2 * kH);
  launch_gemm(stream, cat16, 2 * kH, 0, WoutT, 2 * kH, 0, attn, kH, 0, attn16, kH,
              nullptr, 0, 0, bout, kB * kT, kH, 2 * kH, 1, 1);

  // ---- enc_proj = tanh(enc_out @ Wcopy + bcopy) ----
  launch_gemm(stream, enc_out16, kH, 0, WcopyT, kH, 0, encproj, kH, 0, encproj16, kH,
              nullptr, 0, 0, bcopy, kB * kS, kH, kH, 1, 1);

  // ---- total = exp(attn_out @ Wgen) built directly in d_out; OOV pad 1e-10 ----
  init_oov_kernel<<<cdiv(kB * kT * kOOV, 256), 256, 0, stream>>>(out, kB * kT * kOOV);
  launch_gemm(stream, attn16, kH, 0, WgenT, kH, 0, out, kVO, 0, nullptr, 0,
              nullptr, 0, 0, nullptr, kB * kT, kV, kH, 2, 1);

  // ---- copy scores + deterministic scatter-add into total ----
  launch_gemm(stream, attn16, kH, (long long)kT * kH, encproj16, kH, (long long)kS * kH,
              cs, kS, (long long)kT * kS, nullptr, 0, nullptr, 0, 0, nullptr,
              kT, kS, kH, 0, kB);
  copy_scatter_kernel<<<cdiv(kB * kT, 256), 256, 0, stream>>>(cs, xoov, xlen, out);

  // ---- log-normalize rows in place ----
  finalize_kernel<<<kB * kT, 256, 0, stream>>>(out);

  // ---- append sh, sc ----
  long long obase = (long long)kB * kT * kVO;
  copy_f32_kernel<<<cdiv(kB * kH, 256), 256, 0, stream>>>(h_dec, out + obase, kB * kH);
  copy_f32_kernel<<<cdiv(kB * kH, 256), 256, 0, stream>>>(c_dec, out + obase + kB * kH, kB * kH);
}
